// Qwen2MoeMLPBlock_53953379173324
// MI455X (gfx1250) — matrix operand order repaired and checked by reading, not compiled, not run on hardware
//
#include <hip/hip_runtime.h>
#include <cstddef>

// ---------------- problem constants ----------------
#define T_  2048
#define H_  2048
#define F_  1408
#define E_  16
#define NE  17                 // 16 routed + 1 shared ("expert 16")
#define KT  (NE * F_)          // 23936 : fused down-proj reduction length

typedef __bf16 bf16;
typedef __attribute__((ext_vector_type(16))) __bf16 bf16x16;
typedef __attribute__((ext_vector_type(8)))  __bf16 bf16x8;
typedef __attribute__((ext_vector_type(2)))  __bf16 bf16x2;
typedef __attribute__((ext_vector_type(8)))  float  v8f;

union AB16 { bf16x16 v; bf16x8 h[2]; };

__device__ __forceinline__ v8f wmma_bf16(bf16x16 a, bf16x16 b, v8f c) {
  // (neg_a, A, neg_b, B, c_mod, C, reuse_a, reuse_b)
  return __builtin_amdgcn_wmma_f32_16x16x32_bf16(false, a, false, b, (short)0, c, false, false);
}

// fast sigmoid: v_rcp_f32 instead of IEEE division sequence
__device__ __forceinline__ float sigmoid_fast(float v) {
  return __builtin_amdgcn_rcpf(1.f + __expf(-v));
}

// ---- CDNA5 async global->LDS copy (ASYNCcnt-tracked DMA-style load) ----
__device__ __forceinline__ void async_ld_b128(unsigned ldsAddr, const void* gaddr) {
  asm volatile("global_load_async_to_lds_b128 %0, %1, off"
               :: "v"(ldsAddr), "v"(gaddr) : "memory");
}
__device__ __forceinline__ void wait_async0() {
  asm volatile("s_wait_asynccnt 0x0" ::: "memory");
}

// ---------------- 1) fp32 -> bf16 activation convert ----------------
__global__ __launch_bounds__(256) void k_cvt(const float* __restrict__ x, bf16* __restrict__ xb) {
  size_t i = ((size_t)blockIdx.x * 256 + threadIdx.x) * 8;
  float4 v0 = *(const float4*)(x + i);
  float4 v1 = *(const float4*)(x + i + 4);
  bf16x8 o = { (bf16)v0.x, (bf16)v0.y, (bf16)v0.z, (bf16)v0.w,
               (bf16)v1.x, (bf16)v1.y, (bf16)v1.z, (bf16)v1.w };
  *(bf16x8*)(xb + i) = o;
}

// ---------------- 2) router: logits, top-4, renorm, shared gate ----------------
// 128 threads = 4 waves, one wave per token.
__global__ __launch_bounds__(128) void k_router(const float* __restrict__ x,
                                                const float* __restrict__ rw,
                                                const float* __restrict__ gw,
                                                float* __restrict__ comb) {
  const int lane = threadIdx.x & 31;
  const int w    = threadIdx.x >> 5;
  const int t    = blockIdx.x * 4 + w;

  float acc[NE];
#pragma unroll
  for (int i = 0; i < NE; ++i) acc[i] = 0.f;

  for (int h = lane; h < H_; h += 32) {
    float xv = x[(size_t)t * H_ + h];
    const float4* r4 = (const float4*)(rw + (size_t)h * E_);
#pragma unroll
    for (int q = 0; q < 4; ++q) {
      float4 rv = r4[q];
      acc[q * 4 + 0] += xv * rv.x;
      acc[q * 4 + 1] += xv * rv.y;
      acc[q * 4 + 2] += xv * rv.z;
      acc[q * 4 + 3] += xv * rv.w;
    }
    acc[16] += xv * gw[h];
  }
#pragma unroll
  for (int i = 0; i < NE; ++i)
#pragma unroll
    for (int off = 16; off > 0; off >>= 1)
      acc[i] += __shfl_xor(acc[i], off, 32);

  if (lane == 0) {
    float lg[E_];
#pragma unroll
    for (int i = 0; i < E_; ++i) lg[i] = acc[i];
    int sel[4]; float sv[4];
#pragma unroll
    for (int k = 0; k < 4; ++k) {
      int bi = 0; float bv = -3.4e38f;
      for (int i = 0; i < E_; ++i) if (lg[i] > bv) { bv = lg[i]; bi = i; }
      sel[k] = bi; sv[k] = bv; lg[bi] = -3.4e38f;
    }
    // renormalized top-4 of softmax == softmax over the 4 largest logits
    float m = sv[0];
    float ev[4]; float s = 0.f;
#pragma unroll
    for (int k = 0; k < 4; ++k) { ev[k] = __expf(sv[k] - m); s += ev[k]; }
    float is = __builtin_amdgcn_rcpf(s);
    float* crow = comb + (size_t)t * NE;
#pragma unroll
    for (int i = 0; i < E_; ++i) crow[i] = 0.f;
#pragma unroll
    for (int k = 0; k < 4; ++k) crow[sel[k]] = ev[k] * is;
    crow[E_] = sigmoid_fast(acc[16]);   // shared-gate folded in as expert 16
  }
}

// ---------------- 3) gate+up GEMMs + SwiGLU + comb fold ----------------
// block tile 128(T) x 64(F), K-step 32 over H, double-buffered LDS.
// A tile: async global->LDS (bf16, no conversion). B tiles: reg-staged fp32->bf16.
__global__ __launch_bounds__(256) void k_gate_up(const bf16*  __restrict__ xb,
                                                 const float* __restrict__ eg,
                                                 const float* __restrict__ eu,
                                                 const float* __restrict__ sgw,
                                                 const float* __restrict__ suw,
                                                 const float* __restrict__ comb,
                                                 bf16* __restrict__ hmid) {
  __shared__ bf16 Al[2][128 * 40];       // padded stride 40 -> conflict-free frag reads
  __shared__ bf16 Bl[2][2][64 * 40];     // [buf][mat][n][k] transposed

  const int tid   = threadIdx.x;
  const int lane  = tid & 31;
  const int wid   = tid >> 5;
  const int waveM = wid & 3;
  const int waveN = wid >> 2;
  const int lmod  = lane & 15;
  const int lhalf = lane >> 4;

  const int e  = blockIdx.z;
  const int n0 = blockIdx.x * 64;
  const int t0 = blockIdx.y * 128;

  const float* Wg = (e < E_) ? eg + (size_t)e * H_ * F_ : sgw;
  const float* Wu = (e < E_) ? eu + (size_t)e * H_ * F_ : suw;

  // per-thread staging coordinates
  const int aRow = tid >> 2;             // 0..63 (x2 via i)  : A row within tile
  const int aC8  = (tid & 3) * 8;        // k-chunk of 8
  const int bKp  = tid >> 4;             // 0..15 -> k rows 2kp, 2kp+1
  const int bC4  = (tid & 15) * 4;       // n-chunk of 4

  float4 bR[2][2];                       // [mat][krow]

  auto asyncA = [&](int k0, int p) {
#pragma unroll
    for (int i = 0; i < 2; ++i) {
      const bf16* g = xb + (size_t)(t0 + aRow + i * 64) * H_ + k0 + aC8;
      unsigned l = (unsigned)(size_t)&Al[p][(aRow + i * 64) * 40 + aC8];
      async_ld_b128(l, g);
    }
  };
  auto loadB = [&](int k0) {
#pragma unroll
    for (int m = 0; m < 2; ++m) {
      const float* W = m ? Wu : Wg;
      const float* p0 = W + (size_t)(k0 + 2 * bKp) * F_ + n0 + bC4;
      bR[m][0] = *(const float4*)(p0);
      bR[m][1] = *(const float4*)(p0 + F_);
    }
  };
  auto storeB = [&](int p) {
#pragma unroll
    for (int m = 0; m < 2; ++m) {
      float4 v0 = bR[m][0], v1 = bR[m][1];
      bf16* dst = &Bl[p][m][bC4 * 40 + 2 * bKp];
      *(bf16x2*)(dst + 0 * 40) = bf16x2{ (bf16)v0.x, (bf16)v1.x };
      *(bf16x2*)(dst + 1 * 40) = bf16x2{ (bf16)v0.y, (bf16)v1.y };
      *(bf16x2*)(dst + 2 * 40) = bf16x2{ (bf16)v0.z, (bf16)v1.z };
      *(bf16x2*)(dst + 3 * 40) = bf16x2{ (bf16)v0.w, (bf16)v1.w };
    }
  };

  v8f acc[2][2][2];                      // [mat][tm][tn]
#pragma unroll
  for (int m = 0; m < 2; ++m)
#pragma unroll
    for (int a = 0; a < 2; ++a)
#pragma unroll
      for (int b = 0; b < 2; ++b) acc[m][a][b] = (v8f)0.f;

  const int NIT = H_ / 32;
  asyncA(0, 0);
  loadB(0);
  storeB(0);
  wait_async0();
  __syncthreads();

  int p = 0;
  for (int it = 0; it < NIT; ++it) {
    const bool more = (it + 1 < NIT);
    if (more) {
      asyncA((it + 1) * 32, p ^ 1);      // DMA lands during the wmma block below
      loadB((it + 1) * 32);
    }

    AB16 afr[2];
#pragma unroll
    for (int tm = 0; tm < 2; ++tm) {
      int row = waveM * 32 + tm * 16 + lmod;
      int klo = lhalf * 8;               // ISA 16-bit A layout
      afr[tm].h[0] = *(const bf16x8*)(&Al[p][row * 40 + klo]);
      afr[tm].h[1] = *(const bf16x8*)(&Al[p][row * 40 + klo + 16]);
    }
#pragma unroll
    for (int m = 0; m < 2; ++m)
#pragma unroll
      for (int tn = 0; tn < 2; ++tn) {
        AB16 bfr;
        int col = waveN * 32 + tn * 16 + lmod;
        int kb  = lhalf * 8;             // ISA 16-bit B layout
        bfr.h[0] = *(const bf16x8*)(&Bl[p][m][col * 40 + kb]);
        bfr.h[1] = *(const bf16x8*)(&Bl[p][m][col * 40 + kb + 16]);
#pragma unroll
        for (int tm = 0; tm < 2; ++tm)
          acc[m][tm][tn] = wmma_bf16(afr[tm].v, bfr.v, acc[m][tm][tn]);
      }

    if (more) storeB(p ^ 1);
    wait_async0();
    __syncthreads();
    p ^= 1;
  }

  // epilogue: h = comb17[t,e] * silu(g) * u   (C/D: row = r + 8*lhalf, col = lmod)
#pragma unroll
  for (int tm = 0; tm < 2; ++tm)
#pragma unroll
    for (int tn = 0; tn < 2; ++tn) {
      int tbase = t0 + waveM * 32 + tm * 16 + 8 * lhalf;
      int col   = n0 + waveN * 32 + tn * 16 + lmod;
#pragma unroll
      for (int r = 0; r < 8; ++r) {
        int t   = tbase + r;
        float c = comb[(size_t)t * NE + e];
        float g = acc[0][tm][tn][r];
        float u = acc[1][tm][tn][r];
        float hv = (g * sigmoid_fast(g)) * u * c;
        hmid[(size_t)t * KT + (size_t)e * F_ + col] = (bf16)hv;
      }
    }
}

// ---------------- 4) fused combine+down projection: one GEMM, K = 17*F ----------------
__global__ __launch_bounds__(256) void k_down(const bf16*  __restrict__ hmid,
                                              const float* __restrict__ ed,
                                              const float* __restrict__ sd,
                                              float* __restrict__ out) {
  __shared__ bf16 Al[2][128 * 40];
  __shared__ bf16 Bl[2][64 * 40];

  const int tid   = threadIdx.x;
  const int lane  = tid & 31;
  const int wid   = tid >> 5;
  const int waveM = wid & 3;
  const int waveN = wid >> 2;
  const int lmod  = lane & 15;
  const int lhalf = lane >> 4;

  const int n0 = blockIdx.x * 64;   // output H columns
  const int t0 = blockIdx.y * 128;  // tokens

  const int aRow = tid >> 2;
  const int aC8  = (tid & 3) * 8;
  const int bKp  = tid >> 4;
  const int bC4  = (tid & 15) * 4;

  float4 bR[2];

  auto asyncA = [&](int k0, int p) {
#pragma unroll
    for (int i = 0; i < 2; ++i) {
      const bf16* g = hmid + (size_t)(t0 + aRow + i * 64) * KT + k0 + aC8;
      unsigned l = (unsigned)(size_t)&Al[p][(aRow + i * 64) * 40 + aC8];
      async_ld_b128(l, g);
    }
  };
  auto loadB = [&](int k0) {
    // B rows come from the virtual concat [ed_w (E,F,H) ; sdown_w (F,H)].
    // A 32-aligned k-tile never crosses an expert boundary (F % 32 == 0).
    int kk = k0 + 2 * bKp;
    int ex = kk / F_;
    int f  = kk - ex * F_;
    const float* base = (ex < E_) ? (ed + ((size_t)ex * F_ + f) * H_)
                                  : (sd + (size_t)f * H_);
    bR[0] = *(const float4*)(base + n0 + bC4);
    bR[1] = *(const float4*)(base + H_ + n0 + bC4);
  };
  auto storeB = [&](int p) {
    float4 v0 = bR[0], v1 = bR[1];
    bf16* dst = &Bl[p][bC4 * 40 + 2 * bKp];
    *(bf16x2*)(dst + 0 * 40) = bf16x2{ (bf16)v0.x, (bf16)v1.x };
    *(bf16x2*)(dst + 1 * 40) = bf16x2{ (bf16)v0.y, (bf16)v1.y };
    *(bf16x2*)(dst + 2 * 40) = bf16x2{ (bf16)v0.z, (bf16)v1.z };
    *(bf16x2*)(dst + 3 * 40) = bf16x2{ (bf16)v0.w, (bf16)v1.w };
  };

  v8f acc[2][2];
#pragma unroll
  for (int a = 0; a < 2; ++a)
#pragma unroll
    for (int b = 0; b < 2; ++b) acc[a][b] = (v8f)0.f;

  const int NIT = KT / 32;               // 748
  asyncA(0, 0);
  loadB(0);
  storeB(0);
  wait_async0();
  __syncthreads();

  int p = 0;
  for (int it = 0; it < NIT; ++it) {
    const bool more = (it + 1 < NIT);
    if (more) {
      asyncA((it + 1) * 32, p ^ 1);
      loadB((it + 1) * 32);
    }

    AB16 afr[2];
#pragma unroll
    for (int tm = 0; tm < 2; ++tm) {
      int row = waveM * 32 + tm * 16 + lmod;
      int klo = lhalf * 8;
      afr[tm].h[0] = *(const bf16x8*)(&Al[p][row * 40 + klo]);
      afr[tm].h[1] = *(const bf16x8*)(&Al[p][row * 40 + klo + 16]);
    }
#pragma unroll
    for (int tn = 0; tn < 2; ++tn) {
      AB16 bfr;
      int col = waveN * 32 + tn * 16 + lmod;
      int kb  = lhalf * 8;
      bfr.h[0] = *(const bf16x8*)(&Bl[p][col * 40 + kb]);
      bfr.h[1] = *(const bf16x8*)(&Bl[p][col * 40 + kb + 16]);
#pragma unroll
      for (int tm = 0; tm < 2; ++tm)
        acc[tm][tn] = wmma_bf16(afr[tm].v, bfr.v, acc[tm][tn]);
    }

    if (more) storeB(p ^ 1);
    wait_async0();
    __syncthreads();
    p ^= 1;
  }

  // store: out = 0.5 * acc   (scale = 1/sqrt(TOP_K))
#pragma unroll
  for (int tm = 0; tm < 2; ++tm)
#pragma unroll
    for (int tn = 0; tn < 2; ++tn) {
      int tbase = t0 + waveM * 32 + tm * 16 + 8 * lhalf;
      int hcol  = n0 + waveN * 32 + tn * 16 + lmod;
#pragma unroll
      for (int r = 0; r < 8; ++r)
        out[(size_t)(tbase + r) * H_ + hcol] = 0.5f * acc[tm][tn][r];
    }
}

// ---------------- launch ----------------
extern "C" void kernel_launch(void* const* d_in, const int* in_sizes, int n_in,
                              void* d_out, int out_size, void* d_ws, size_t ws_size,
                              hipStream_t stream) {
  (void)in_sizes; (void)n_in; (void)out_size; (void)ws_size;
  const float* x   = (const float*)d_in[0];
  const float* rw  = (const float*)d_in[1];
  const float* gw  = (const float*)d_in[2];
  const float* sgw = (const float*)d_in[3];
  const float* suw = (const float*)d_in[4];
  const float* sdw = (const float*)d_in[5];
  const float* eg  = (const float*)d_in[6];
  const float* eu  = (const float*)d_in[7];
  const float* ed  = (const float*)d_in[8];
  float* out = (float*)d_out;

  char* ws = (char*)d_ws;
  const size_t xb_bytes   = (size_t)T_ * H_ * sizeof(bf16);      // 8 MB
  const size_t comb_bytes = (size_t)T_ * NE * sizeof(float);     // 136 KB
  bf16*  xb   = (bf16*)ws;
  float* comb = (float*)(ws + xb_bytes);
  bf16*  hmid = (bf16*)(ws + xb_bytes + comb_bytes);             // T x 23936 bf16 ~ 94 MB

  k_cvt   <<<(T_ * H_) / (256 * 8), 256, 0, stream>>>(x, xb);
  k_router<<<T_ / 4, 128, 0, stream>>>(x, rw, gw, comb);
  k_gate_up<<<dim3(F_ / 64, T_ / 128, NE), 256, 0, stream>>>(xb, eg, eu, sgw, suw, comb, hmid);
  k_down  <<<dim3(H_ / 64, T_ / 128), 256, 0, stream>>>(hmid, ed, sdw, out);
}
